// TernaryLinearSTE_18614388260897
// MI455X (gfx1250) — compile-verified, matrix operand order checked by reading, not run on hardware
//
#include <hip/hip_runtime.h>

typedef __attribute__((ext_vector_type(16))) _Float16 v16h;
typedef __attribute__((ext_vector_type(8)))  _Float16 v8h;
typedef __attribute__((ext_vector_type(8)))  float    v8f;

#define RED_BLOCKS 1024
#define RED_THREADS 256
#define THRESH 0.7f

// ---------------- ws layout ----------------
// scal[0] = sumAbs(|w|)   scal[1] = delta   scal[2] = alpha
// partA/partS/partC [RED_BLOCKS] each
// qw at byte 16384                : ternary weights f16, (N,K) row-major (32 MB)
// xh after qw (256-aligned)       : x converted to f16, (M,K) row-major (64 MB) [if ws fits]

// ---------- pass 1: sum |w| (deterministic grid-stride + tree) ----------
__global__ __launch_bounds__(RED_THREADS) void kAbsSum(const float* __restrict__ w,
                                                       float* __restrict__ part, int n) {
    __shared__ float s[RED_THREADS];
    float acc = 0.f;
    for (int i = blockIdx.x * blockDim.x + threadIdx.x; i < n; i += gridDim.x * blockDim.x)
        acc += fabsf(w[i]);
    s[threadIdx.x] = acc;
    __syncthreads();
    for (int off = RED_THREADS / 2; off > 0; off >>= 1) {
        if (threadIdx.x < off) s[threadIdx.x] += s[threadIdx.x + off];
        __syncthreads();
    }
    if (threadIdx.x == 0) part[blockIdx.x] = s[0];
}

__global__ __launch_bounds__(RED_THREADS) void kFinishDelta(const float* __restrict__ part,
                                                            int nb, float* __restrict__ scal,
                                                            float invN) {
    __shared__ float s[RED_THREADS];
    float acc = 0.f;
    for (int i = threadIdx.x; i < nb; i += RED_THREADS) acc += part[i];
    s[threadIdx.x] = acc;
    __syncthreads();
    for (int off = RED_THREADS / 2; off > 0; off >>= 1) {
        if (threadIdx.x < off) s[threadIdx.x] += s[threadIdx.x + off];
        __syncthreads();
    }
    if (threadIdx.x == 0) {
        scal[0] = s[0];
        scal[1] = THRESH * s[0] * invN;   // delta = 0.7 * mean|w|
    }
}

// ---------- pass 2: ternarize -> f16, partial masked-sum / count ----------
__global__ __launch_bounds__(RED_THREADS) void kTernarize(const float* __restrict__ w,
                                                          _Float16* __restrict__ qw,
                                                          const float* __restrict__ scal,
                                                          float* __restrict__ partS,
                                                          float* __restrict__ partC, int n) {
    __shared__ float ss[RED_THREADS];
    __shared__ float sc[RED_THREADS];
    const float delta = scal[1];
    float ms = 0.f, cnt = 0.f;
    for (int i = blockIdx.x * blockDim.x + threadIdx.x; i < n; i += gridDim.x * blockDim.x) {
        float v = w[i];
        float t = (v > delta) ? 1.f : ((v < -delta) ? -1.f : 0.f);
        qw[i] = (_Float16)t;
        if (t != 0.f) { ms += fabsf(v); cnt += 1.f; }
    }
    ss[threadIdx.x] = ms;
    sc[threadIdx.x] = cnt;
    __syncthreads();
    for (int off = RED_THREADS / 2; off > 0; off >>= 1) {
        if (threadIdx.x < off) {
            ss[threadIdx.x] += ss[threadIdx.x + off];
            sc[threadIdx.x] += sc[threadIdx.x + off];
        }
        __syncthreads();
    }
    if (threadIdx.x == 0) { partS[blockIdx.x] = ss[0]; partC[blockIdx.x] = sc[0]; }
}

__global__ __launch_bounds__(RED_THREADS) void kFinishAlpha(const float* __restrict__ partS,
                                                            const float* __restrict__ partC,
                                                            int nb, float* __restrict__ scal,
                                                            float invN) {
    __shared__ float ss[RED_THREADS];
    __shared__ float sc[RED_THREADS];
    float ms = 0.f, cnt = 0.f;
    for (int i = threadIdx.x; i < nb; i += RED_THREADS) { ms += partS[i]; cnt += partC[i]; }
    ss[threadIdx.x] = ms;
    sc[threadIdx.x] = cnt;
    __syncthreads();
    for (int off = RED_THREADS / 2; off > 0; off >>= 1) {
        if (threadIdx.x < off) {
            ss[threadIdx.x] += ss[threadIdx.x + off];
            sc[threadIdx.x] += sc[threadIdx.x + off];
        }
        __syncthreads();
    }
    if (threadIdx.x == 0)
        scal[2] = (sc[0] > 0.f) ? (ss[0] / sc[0]) : (scal[0] * invN);
}

// ---------- x: f32 -> f16 one-shot conversion (8 elems / thread-iter) ----------
__global__ __launch_bounds__(RED_THREADS) void kCvtX(const float* __restrict__ x,
                                                     _Float16* __restrict__ xh, int n) {
    const int stride = gridDim.x * blockDim.x * 8;
    for (int i = (blockIdx.x * blockDim.x + threadIdx.x) * 8; i < n; i += stride) {
        float4 f0 = *(const float4*)(x + i);
        float4 f1 = *(const float4*)(x + i + 4);
        v8h h;
        h[0] = (_Float16)f0.x; h[1] = (_Float16)f0.y; h[2] = (_Float16)f0.z; h[3] = (_Float16)f0.w;
        h[4] = (_Float16)f1.x; h[5] = (_Float16)f1.y; h[6] = (_Float16)f1.z; h[7] = (_Float16)f1.w;
        *(v8h*)(xh + i) = h;
    }
}

// ---------- WMMA GEMM helpers ----------
__device__ __forceinline__ void wmma_all(v8f acc[4][2], const v16h a[4], const v16h b[2]) {
#pragma unroll
    for (int t = 0; t < 4; ++t)
#pragma unroll
        for (int u = 0; u < 2; ++u)
            acc[t][u] = __builtin_amdgcn_wmma_f32_16x16x32_f16(
                false, a[t], false, b[u], (short)0, acc[t][u], false, false);
}

__device__ __forceinline__ void load_frags_h(const _Float16* const arow[4],
                                             const _Float16* const brow[2],
                                             int k0, v16h a[4], v16h b[2]) {
#pragma unroll
    for (int u = 0; u < 2; ++u) b[u] = *(const v16h*)(brow[u] + k0);
#pragma unroll
    for (int t = 0; t < 4; ++t) a[t] = *(const v16h*)(arow[t] + k0);
}

// ---------- GEMM (fast path): all-f16 operands, register double-buffered K ----------
// Block: 256 threads = 8 waves (2 Mwaves x 4 Nwaves); block tile 128x128.
// Wave tile 64x32 = 4x2 WMMA tiles. A-frag/B-frag: 16 contiguous K-halfs per lane.
__global__ __launch_bounds__(256) void kGemmWmmaF16(const _Float16* __restrict__ xh,
                                                    const _Float16* __restrict__ qw,
                                                    const float* __restrict__ bias,
                                                    const float* __restrict__ scal,
                                                    float* __restrict__ out,
                                                    int M, int N, int K) {
    const int lane = threadIdx.x & 31;
    const int wave = threadIdx.x >> 5;
    const int l16  = lane & 15;
    const int kh   = lane >> 4;
    const int wm   = wave >> 2;
    const int wn   = wave & 3;

    const long m0 = (long)blockIdx.y * 128 + wm * 64;
    const long n0 = (long)blockIdx.x * 128 + wn * 32;

    const _Float16* arow[4];
#pragma unroll
    for (int t = 0; t < 4; ++t)
        arow[t] = xh + (m0 + t * 16 + l16) * (long)K + kh * 16;
    const _Float16* brow[2];
#pragma unroll
    for (int u = 0; u < 2; ++u)
        brow[u] = qw + (n0 + u * 16 + l16) * (long)K + kh * 16;

    v8f acc[4][2] = {};
    v16h a0[4], b0[2], a1[4], b1[2];

    load_frags_h(arow, brow, 0, a0, b0);
    // K % 64 == 0 (K = 4096). Ping-pong: prefetch k+32 / k+64 before consuming k.
    for (int k0 = 0; k0 < K - 64; k0 += 64) {
        load_frags_h(arow, brow, k0 + 32, a1, b1);
        wmma_all(acc, a0, b0);
        load_frags_h(arow, brow, k0 + 64, a0, b0);
        wmma_all(acc, a1, b1);
    }
    load_frags_h(arow, brow, K - 32, a1, b1);
    wmma_all(acc, a0, b0);
    wmma_all(acc, a1, b1);

    const float alpha = scal[2];
#pragma unroll
    for (int u = 0; u < 2; ++u) {
        const long n  = n0 + u * 16 + l16;
        const float bv = bias[n];
#pragma unroll
        for (int t = 0; t < 4; ++t) {
            const long mb = m0 + t * 16 + kh * 8;
#pragma unroll
            for (int i = 0; i < 8; ++i)
                out[(mb + i) * (long)N + n] = acc[t][u][i] * alpha + bv;
        }
    }
}

// ---------- GEMM (fallback, small ws): f32 x with in-loop cvt ----------
__device__ __forceinline__ v16h load_cvt16(const float* __restrict__ p) {
    const float4* q = (const float4*)p;
    float4 f0 = q[0], f1 = q[1], f2 = q[2], f3 = q[3];
    v16h r;
    r[0]  = (_Float16)f0.x; r[1]  = (_Float16)f0.y; r[2]  = (_Float16)f0.z; r[3]  = (_Float16)f0.w;
    r[4]  = (_Float16)f1.x; r[5]  = (_Float16)f1.y; r[6]  = (_Float16)f1.z; r[7]  = (_Float16)f1.w;
    r[8]  = (_Float16)f2.x; r[9]  = (_Float16)f2.y; r[10] = (_Float16)f2.z; r[11] = (_Float16)f2.w;
    r[12] = (_Float16)f3.x; r[13] = (_Float16)f3.y; r[14] = (_Float16)f3.z; r[15] = (_Float16)f3.w;
    return r;
}

__global__ __launch_bounds__(256) void kGemmWmmaF32(const float* __restrict__ x,
                                                    const _Float16* __restrict__ qw,
                                                    const float* __restrict__ bias,
                                                    const float* __restrict__ scal,
                                                    float* __restrict__ out,
                                                    int M, int N, int K) {
    const int lane = threadIdx.x & 31;
    const int wave = threadIdx.x >> 5;
    const int l16  = lane & 15;
    const int kh   = lane >> 4;
    const int wm   = wave >> 2;
    const int wn   = wave & 3;

    const long m0 = (long)blockIdx.y * 128 + wm * 64;
    const long n0 = (long)blockIdx.x * 128 + wn * 32;

    const float* arow[4];
#pragma unroll
    for (int t = 0; t < 4; ++t)
        arow[t] = x + (m0 + t * 16 + l16) * (long)K + kh * 16;
    const _Float16* brow[2];
#pragma unroll
    for (int u = 0; u < 2; ++u)
        brow[u] = qw + (n0 + u * 16 + l16) * (long)K + kh * 16;

    v8f acc[4][2] = {};
    for (int k0 = 0; k0 < K; k0 += 32) {
        v16h b[2];
#pragma unroll
        for (int u = 0; u < 2; ++u) b[u] = *(const v16h*)(brow[u] + k0);
        v16h a[4];
#pragma unroll
        for (int t = 0; t < 4; ++t) a[t] = load_cvt16(arow[t] + k0);
        wmma_all(acc, a, b);
    }

    const float alpha = scal[2];
#pragma unroll
    for (int u = 0; u < 2; ++u) {
        const long n  = n0 + u * 16 + l16;
        const float bv = bias[n];
#pragma unroll
        for (int t = 0; t < 4; ++t) {
            const long mb = m0 + t * 16 + kh * 8;
#pragma unroll
            for (int i = 0; i < 8; ++i)
                out[(mb + i) * (long)N + n] = acc[t][u][i] * alpha + bv;
        }
    }
}

extern "C" void kernel_launch(void* const* d_in, const int* in_sizes, int n_in,
                              void* d_out, int out_size, void* d_ws, size_t ws_size,
                              hipStream_t stream) {
    const float* x    = (const float*)d_in[0];
    const float* w    = (const float*)d_in[1];
    const float* bias = (const float*)d_in[2];
    float*       out  = (float*)d_out;

    const int N  = in_sizes[2];            // 4096
    const int K  = in_sizes[1] / N;        // 4096
    const int M  = in_sizes[0] / K;        // 8192
    const int nW = in_sizes[1];
    const int nX = in_sizes[0];

    float*    scal  = (float*)d_ws;
    float*    partA = scal + 16;
    float*    partS = partA + RED_BLOCKS;
    float*    partC = partS + RED_BLOCKS;

    const size_t qwOff   = 16384;
    const size_t qwBytes = (size_t)nW * 2;
    const size_t xhOff   = (qwOff + qwBytes + 255) & ~(size_t)255;
    const size_t xhBytes = (size_t)nX * 2;

    _Float16* qw = (_Float16*)((char*)d_ws + qwOff);
    _Float16* xh = (_Float16*)((char*)d_ws + xhOff);
    const bool fast = (ws_size >= xhOff + xhBytes);   // ws_size constant -> deterministic

    const float invN = 1.0f / (float)nW;

    kAbsSum<<<RED_BLOCKS, RED_THREADS, 0, stream>>>(w, partA, nW);
    kFinishDelta<<<1, RED_THREADS, 0, stream>>>(partA, RED_BLOCKS, scal, invN);
    kTernarize<<<RED_BLOCKS, RED_THREADS, 0, stream>>>(w, qw, scal, partS, partC, nW);
    kFinishAlpha<<<1, RED_THREADS, 0, stream>>>(partS, partC, RED_BLOCKS, scal, invN);

    dim3 grid(N / 128, M / 128);
    if (fast) {
        kCvtX<<<2048, RED_THREADS, 0, stream>>>(x, xh, nX);
        kGemmWmmaF16<<<grid, 256, 0, stream>>>(xh, qw, bias, scal, out, M, N, K);
    } else {
        kGemmWmmaF32<<<grid, 256, 0, stream>>>(x, qw, bias, scal, out, M, N, K);
    }
}
